// DifferentialAttention_29978871726251
// MI455X (gfx1250) — matrix operand order repaired and checked by reading, not compiled, not run on hardware
//
#include <hip/hip_runtime.h>

// ---------------------------------------------------------------------------
// DifferentialAttention on MI455X (gfx1250, wave32, WMMA).
// All heavy math runs through v_wmma_f32_16x16x32_f16.
// Flash-style attention: no S x S score tensor ever hits memory.
// Tile staging uses the CDNA5 async global->LDS path (ASYNCcnt).
// ---------------------------------------------------------------------------

#define B_ 2
#define S_ 2048
#define D_ 1024
#define H_ 16
#define DH_ 64
#define SCALE_ 0.125f   // DH^-0.5

typedef _Float16 half_t;
typedef __attribute__((ext_vector_type(16))) _Float16 v16h;
typedef __attribute__((ext_vector_type(8)))  _Float16 v8h;
typedef __attribute__((ext_vector_type(8)))  float    v8f;

static __device__ __forceinline__ v16h cat16(v8h lo, v8h hi) {
  v16h r;
#pragma unroll
  for (int i = 0; i < 8; ++i) { r[i] = lo[i]; r[i + 8] = hi[i]; }
  return r;
}

// D = A(16x32 f16) * B(32x16 f16) + C(16x16 f32)
static __device__ __forceinline__ v8f wmma16x16x32(v16h a, v16h b, v8f c) {
  return __builtin_amdgcn_wmma_f32_16x16x32_f16(
      /*neg_a=*/false, a, /*neg_b=*/false, b,
      /*c_mod=*/(short)0, c, /*reuse_a=*/false, /*reuse_b=*/false);
}

// Per-lane 16B async DMA: global -> LDS, tracked by ASYNCcnt (ISA 08 §4).
// LDS byte address = low 32 bits of the flat shared pointer (LDS aperture
// maps addr[31:0] to the LDS offset).
static __device__ __forceinline__ void async_copy_b128(void* lds_dst,
                                                       const void* gsrc) {
  asm volatile("global_load_async_to_lds_b128 %0, %1, off"
               :
               : "v"((unsigned)(unsigned long long)lds_dst), "v"(gsrc)
               : "memory");
}
static __device__ __forceinline__ void wait_asynccnt0() {
  asm volatile("s_wait_asynccnt 0x0" ::: "memory");
}

// ---------------------------------------------------------------------------
// fp32 -> f16 convert (grid-stride)
// ---------------------------------------------------------------------------
__global__ void cvt_f32_f16(const float* __restrict__ src,
                            half_t* __restrict__ dst, int n) {
  for (int i = blockIdx.x * blockDim.x + threadIdx.x; i < n;
       i += gridDim.x * blockDim.x)
    dst[i] = (half_t)src[i];
}

// ---------------------------------------------------------------------------
// C[M,N](OutT) = A[M,K](f16, row-major) * W[K,N](f16, row-major) + bias[N]
// Block: 256 threads = 8 waves. Block tile 128x64; each wave: 16x64 strip.
// W tile (32x64) is async-DMA'd into a double-buffered LDS tile shared by all
// 8 waves (removes 8x redundant global traffic); A-frag loads straight from
// global (each wave owns distinct rows, two contiguous 16B chunks per lane).
// ---------------------------------------------------------------------------
template <typename OutT>
__global__ void __launch_bounds__(256)
gemm_bias(const half_t* __restrict__ A, const half_t* __restrict__ W,
          const float* __restrict__ bias, OutT* __restrict__ C,
          int M, int N, int K) {
  __shared__ __align__(32) half_t wlds[2][32 * 64];   // double-buffered B tile

  const int tid  = threadIdx.x;
  const int lane = tid & 31;
  const int wave = tid >> 5;
  const int ln   = lane & 15;
  const int hf   = lane >> 4;
  const int rowBase = blockIdx.x * 128 + wave * 16;
  const int colBase = blockIdx.y * 64;

  const v8f vzero = {0.f, 0.f, 0.f, 0.f, 0.f, 0.f, 0.f, 0.f};
  v8f acc[4];
#pragma unroll
  for (int t = 0; t < 4; ++t) acc[t] = vzero;

  const half_t* arow = A + (size_t)(rowBase + ln) * K;

  // 256 threads x 16B = one 32x64 f16 tile per stage call
  const int skr = tid >> 3;          // W row within tile 0..31
  const int sd0 = (tid & 7) * 8;     // col offset 0..56

  // prologue: stage first tile
  async_copy_b128(&wlds[0][skr * 64 + sd0],
                  W + (size_t)skr * N + colBase + sd0);
  wait_asynccnt0();
  __syncthreads();

  int cur = 0;
  for (int k = 0; k < K; k += 32) {
    if (k + 32 < K)   // uniform branch: prefetch next tile into other buffer
      async_copy_b128(&wlds[cur ^ 1][skr * 64 + sd0],
                      W + (size_t)(k + 32 + skr) * N + colBase + sd0);

    __builtin_prefetch(arow + k + 128, 0, 1);   // global_prefetch_b8
    v8h a0 = *(const v8h*)(arow + k + hf * 8);
    v8h a1 = *(const v8h*)(arow + k + hf * 8 + 16);
    v16h af = cat16(a0, a1);

    const half_t* wl = &wlds[cur][ln];  // B-frag column N=lane, K in table order
#pragma unroll
    for (int t = 0; t < 4; ++t) {
      v16h bf;
#pragma unroll
      for (int s = 0; s < 16; ++s)
        bf[s] = wl[(hf * 8 + (s & 7) + (s >> 3) * 16) * 64 + t * 16];
      acc[t] = wmma16x16x32(af, bf, acc[t]);
    }

    wait_asynccnt0();
    __syncthreads();
    cur ^= 1;
  }

#pragma unroll
  for (int t = 0; t < 4; ++t) {
    const int col = colBase + t * 16 + ln;
    const float bv = bias[col];
#pragma unroll
    for (int i = 0; i < 8; ++i) {
      const int row = rowBase + hf * 8 + i;
      C[(size_t)row * N + col] = (OutT)(acc[t][i] + bv);
    }
  }
}

// ---------------------------------------------------------------------------
// Differential flash attention.
// Grid: B*H*(S/64) blocks, 128 threads = 4 waves; wave owns 16 query rows.
// Per k-tile (32 keys): async-DMA V (32x64, natural layout) into LDS,
// transpose-stage K1^T,K2^T (64x32) via VGPRs, 8 WMMAs for s1/s2, online
// softmax (two accumulator sets), repack P via per-wave LDS scratch,
// 8 WMMAs for O1/O2 += P x V.
// ---------------------------------------------------------------------------
__global__ void __launch_bounds__(128)
diff_attn(const half_t* __restrict__ Q, const half_t* __restrict__ K,
          const half_t* __restrict__ V, const float* __restrict__ mask,
          const float* __restrict__ lam_p, half_t* __restrict__ O) {
  __shared__ __align__(32) half_t kt1[DH_ * 32];      // [d][k]
  __shared__ __align__(32) half_t kt2[DH_ * 32];      // [d][k]
  __shared__ __align__(32) half_t vtile[32 * DH_];    // [k][d]
  __shared__ __align__(32) half_t ps[4][2][16 * 32];  // per-wave P repack

  const int tid  = threadIdx.x;
  const int lane = tid & 31;
  const int wave = tid >> 5;
  const int ln   = lane & 15;
  const int hf   = lane >> 4;

  const int blk = blockIdx.x;
  const int qt  = blk & (S_ / 64 - 1);
  const int h   = (blk >> 5) & (H_ - 1);
  const int b   = blk >> 9;

  const float lam = lam_p[0];
  const int tok0  = b * S_;

  // Q fragments (held in VGPRs for the whole key loop)
  const int qtok = tok0 + qt * 64 + wave * 16 + ln;
  const half_t* q1p = Q + (size_t)qtok * (2 * D_) + h * DH_;
  const half_t* q2p = q1p + D_;
  v16h q1f[2], q2f[2];
#pragma unroll
  for (int p = 0; p < 2; ++p) {
    const int off = p * 32 + hf * 8;
    q1f[p] = cat16(*(const v8h*)(q1p + off), *(const v8h*)(q1p + off + 16));
    q2f[p] = cat16(*(const v8h*)(q2p + off), *(const v8h*)(q2p + off + 16));
  }

  const v8f vzero = {0.f, 0.f, 0.f, 0.f, 0.f, 0.f, 0.f, 0.f};
  float m1[8], l1[8], m2[8], l2[8], c1[8], c2[8];
  v8f o1[4], o2[4];
#pragma unroll
  for (int i = 0; i < 8; ++i) { m1[i] = m2[i] = -1e30f; l1[i] = l2[i] = 0.f; }
#pragma unroll
  for (int t = 0; t < 4; ++t) { o1[t] = vzero; o2[t] = vzero; }

  for (int kt = 0; kt < S_ / 32; ++kt) {
    __syncthreads();
    // ---- cooperative staging: V via async DMA; K1^T,K2^T transposed ----
#pragma unroll
    for (int j = 0; j < 2; ++j) {
      const int c  = tid + j * 128;   // 256 chunks of 8 halfs
      const int kr = c >> 3;          // key row in tile 0..31
      const int d0 = (c & 7) * 8;     // d offset 0..56
      const half_t* vrow = V + (size_t)(tok0 + kt * 32 + kr) * D_ + h * DH_;
      async_copy_b128(vtile + kr * DH_ + d0, vrow + d0);

      const half_t* krow =
          K + (size_t)(tok0 + kt * 32 + kr) * (2 * D_) + h * DH_;
      v8h k1v = *(const v8h*)(krow + d0);
      v8h k2v = *(const v8h*)(krow + D_ + d0);
#pragma unroll
      for (int i = 0; i < 8; ++i) {
        kt1[(d0 + i) * 32 + kr] = k1v[i];
        kt2[(d0 + i) * 32 + kr] = k2v[i];
      }
    }
    wait_asynccnt0();
    __syncthreads();

    // ---- scores: 16x32 tile as two 16x16 halves, both branches ----
    v8f s1[2], s2[2];
#pragma unroll
    for (int nh = 0; nh < 2; ++nh) {
      s1[nh] = vzero;
      s2[nh] = vzero;
#pragma unroll
      for (int p = 0; p < 2; ++p) {
        v16h b1, b2;   // B-frag: column N=lane (key), K (d) in table order
#pragma unroll
        for (int s = 0; s < 16; ++s) {
          const int d = p * 32 + hf * 8 + (s & 7) + (s >> 3) * 16;
          b1[s] = kt1[d * 32 + nh * 16 + ln];
          b2[s] = kt2[d * 32 + nh * 16 + ln];
        }
        s1[nh] = wmma16x16x32(q1f[p], b1, s1[nh]);
        s2[nh] = wmma16x16x32(q2f[p], b2, s2[nh]);
      }
      const float bias =
          (1.0f - mask[tok0 + kt * 32 + nh * 16 + ln]) * -10000.0f;
#pragma unroll
      for (int i = 0; i < 8; ++i) {
        s1[nh][i] = s1[nh][i] * SCALE_ + bias;
        s2[nh][i] = s2[nh][i] * SCALE_ + bias;
      }
    }

    // ---- online softmax (row = hf*8+i, spread over 16 lanes of a half) ----
#pragma unroll
    for (int i = 0; i < 8; ++i) {
      float mx1 = fmaxf(s1[0][i], s1[1][i]);
      float mx2 = fmaxf(s2[0][i], s2[1][i]);
#pragma unroll
      for (int mm = 8; mm >= 1; mm >>= 1) {
        mx1 = fmaxf(mx1, __shfl_xor(mx1, mm, 32));
        mx2 = fmaxf(mx2, __shfl_xor(mx2, mm, 32));
      }
      const float n1 = fmaxf(m1[i], mx1);
      const float n2 = fmaxf(m2[i], mx2);
      c1[i] = __expf(m1[i] - n1);
      c2[i] = __expf(m2[i] - n2);
      float p10 = __expf(s1[0][i] - n1), p11 = __expf(s1[1][i] - n1);
      float p20 = __expf(s2[0][i] - n2), p21 = __expf(s2[1][i] - n2);
      float r1 = p10 + p11, r2 = p20 + p21;
#pragma unroll
      for (int mm = 8; mm >= 1; mm >>= 1) {
        r1 += __shfl_xor(r1, mm, 32);
        r2 += __shfl_xor(r2, mm, 32);
      }
      l1[i] = l1[i] * c1[i] + r1;
      l2[i] = l2[i] * c2[i] + r2;
      m1[i] = n1;
      m2[i] = n2;
      s1[0][i] = p10; s1[1][i] = p11;
      s2[0][i] = p20; s2[1][i] = p21;
    }

    // ---- repack P: C layout -> A-fragment layout via per-wave LDS ----
#pragma unroll
    for (int nh = 0; nh < 2; ++nh)
#pragma unroll
      for (int i = 0; i < 8; ++i) {
        ps[wave][0][(hf * 8 + i) * 32 + nh * 16 + ln] = (half_t)s1[nh][i];
        ps[wave][1][(hf * 8 + i) * 32 + nh * 16 + ln] = (half_t)s2[nh][i];
      }
    __syncthreads();

    v16h p1f = cat16(*(const v8h*)(&ps[wave][0][ln * 32 + hf * 8]),
                     *(const v8h*)(&ps[wave][0][ln * 32 + hf * 8 + 16]));
    v16h p2f = cat16(*(const v8h*)(&ps[wave][1][ln * 32 + hf * 8]),
                     *(const v8h*)(&ps[wave][1][ln * 32 + hf * 8 + 16]));

    // ---- rescale and accumulate O += P x V ----
    const int kk = hf * 8;  // B-frag: column N=lane (d), K (key) in table order
#pragma unroll
    for (int t = 0; t < 4; ++t) {
#pragma unroll
      for (int i = 0; i < 8; ++i) { o1[t][i] *= c1[i]; o2[t][i] *= c2[i]; }
      v16h vf;
#pragma unroll
      for (int s = 0; s < 16; ++s)
        vf[s] = vtile[(kk + (s & 7) + (s >> 3) * 16) * DH_ + t * 16 + ln];
      o1[t] = wmma16x16x32(p1f, vf, o1[t]);
      o2[t] = wmma16x16x32(p2f, vf, o2[t]);
    }
  }

  // ---- epilogue: O1/l1 - lam*O2/l2, write f16 head output ----
#pragma unroll
  for (int t = 0; t < 4; ++t) {
#pragma unroll
    for (int i = 0; i < 8; ++i) {
      const float r = o1[t][i] / l1[i] - lam * (o2[t][i] / l2[i]);
      const int tok = tok0 + qt * 64 + wave * 16 + hf * 8 + i;
      const int col = h * DH_ + t * 16 + ln;
      O[(size_t)tok * D_ + col] = (half_t)r;
    }
  }
}

// ---------------------------------------------------------------------------
extern "C" void kernel_launch(void* const* d_in, const int* in_sizes, int n_in,
                              void* d_out, int out_size, void* d_ws,
                              size_t ws_size, hipStream_t stream) {
  (void)in_sizes; (void)n_in; (void)out_size; (void)ws_size;
  const float* hidden = (const float*)d_in[0];
  const float* mask   = (const float*)d_in[1];
  const float* Wq = (const float*)d_in[2];
  const float* bq = (const float*)d_in[3];
  const float* Wk = (const float*)d_in[4];
  const float* bk = (const float*)d_in[5];
  const float* Wv = (const float*)d_in[6];
  const float* bv = (const float*)d_in[7];
  const float* Wo = (const float*)d_in[8];
  const float* bo = (const float*)d_in[9];
  const float* lam = (const float*)d_in[10];
  float* out = (float*)d_out;

  // Workspace layout (f16), ~68 MB total.
  char* w = (char*)d_ws;
  size_t o = 0;
  half_t* Xh  = (half_t*)(w + o); o += (size_t)B_ * S_ * D_ * 2;
  half_t* Wqh = (half_t*)(w + o); o += (size_t)D_ * 2 * D_ * 2;
  half_t* Wkh = (half_t*)(w + o); o += (size_t)D_ * 2 * D_ * 2;
  half_t* Wvh = (half_t*)(w + o); o += (size_t)D_ * D_ * 2;
  half_t* Woh = (half_t*)(w + o); o += (size_t)D_ * D_ * 2;
  half_t* Qh  = (half_t*)(w + o); o += (size_t)B_ * S_ * 2 * D_ * 2;
  half_t* Kh  = (half_t*)(w + o); o += (size_t)B_ * S_ * 2 * D_ * 2;
  half_t* Vh  = (half_t*)(w + o); o += (size_t)B_ * S_ * D_ * 2;
  half_t* AOh = (half_t*)(w + o); o += (size_t)B_ * S_ * D_ * 2;

  // 1) convert fp32 -> f16 (weights stay row-major; WMMA B-frag reads rows)
  cvt_f32_f16<<<2048, 256, 0, stream>>>(hidden, Xh, B_ * S_ * D_);
  cvt_f32_f16<<<2048, 256, 0, stream>>>(Wq, Wqh, D_ * 2 * D_);
  cvt_f32_f16<<<2048, 256, 0, stream>>>(Wk, Wkh, D_ * 2 * D_);
  cvt_f32_f16<<<1024, 256, 0, stream>>>(Wv, Wvh, D_ * D_);
  cvt_f32_f16<<<1024, 256, 0, stream>>>(Wo, Woh, D_ * D_);

  // 2) QKV projections (M=4096, K=1024)
  gemm_bias<half_t><<<dim3(32, 32), 256, 0, stream>>>(Xh, Wqh, bq, Qh,
                                                      B_ * S_, 2 * D_, D_);
  gemm_bias<half_t><<<dim3(32, 32), 256, 0, stream>>>(Xh, Wkh, bk, Kh,
                                                      B_ * S_, 2 * D_, D_);
  gemm_bias<half_t><<<dim3(32, 16), 256, 0, stream>>>(Xh, Wvh, bv, Vh,
                                                      B_ * S_, D_, D_);

  // 3) differential flash attention
  diff_attn<<<B_ * H_ * (S_ / 64), 128, 0, stream>>>(Qh, Kh, Vh, mask, lam,
                                                     AOh);

  // 4) output projection -> fp32 result
  gemm_bias<float><<<dim3(32, 16), 256, 0, stream>>>(AOh, Woh, bo, out,
                                                     B_ * S_, D_, D_);
}
